// TokenKNN_87986700026091
// MI455X (gfx1250) — matrix operand order repaired and checked by reading, not compiled, not run on hardware
//
#include <hip/hip_runtime.h>
#include <hip/hip_bf16.h>

// ---------------- problem constants ----------------
#define V_SIZE   128000
#define D_DIM    2048
#define N_Q      1024
#define TOPK     5

// ---------------- tiling ----------------
#define COLS_PER_WG  64          // vocab columns per workgroup (4 WMMA tiles of 16)
#define NTILES       4
#define WAVES        8           // 256 threads / wave32
#define M_PER_WAVE   128         // queries per wave (8 m-blocks of 16)
#define MBLOCKS      8
#define KCHUNK       256         // K slice staged in LDS per iteration
#define KSTEPS       (KCHUNK / 32)
#define LDS_STRIDE   (KCHUNK + 8)   // bf16 elems; +16B pad per row kills bank conflicts

// LDS: tile phase = 64 * 264 * 2 = 33792 B ; merge phase = 64*16*5*(4+4) = 40960 B
#define LDS_BYTES 40960

typedef __attribute__((ext_vector_type(16))) __bf16 v16bf;
typedef __attribute__((ext_vector_type(8)))  __bf16 v8bf;
typedef __attribute__((ext_vector_type(8)))  float  v8f;

__device__ __forceinline__ unsigned int pack_bf16x2(float lo, float hi) {
  unsigned int ul = __float_as_uint(lo);
  unsigned int uh = __float_as_uint(hi);
  ul = (ul + 0x7FFFu + ((ul >> 16) & 1u)) >> 16;                 // RNE truncate
  uh = (uh + 0x7FFFu + ((uh >> 16) & 1u)) & 0xFFFF0000u;
  return ul | uh;
}

// branchless descending top-5 insert
__device__ __forceinline__ void insert5(float* v, int* ix, float nv, int ni) {
#pragma unroll
  for (int s = 0; s < TOPK; ++s) {
    bool gt = nv > v[s];
    float ov = v[s]; int oi = ix[s];
    v[s]  = gt ? nv : ov;
    ix[s] = gt ? ni : oi;
    nv = gt ? ov : nv;
    ni = gt ? oi : ni;
  }
}

// ---------------------------------------------------------------------------
// Kernel 1: gather query rows, L2-normalize (x / max(||x||,eps)), emit bf16 Q.
// One block per query row; thread t owns 8 contiguous floats -> 8 bf16.
// ---------------------------------------------------------------------------
__global__ void __launch_bounds__(256)
tk_gather_normalize(const float* __restrict__ emb, const int* __restrict__ tok,
                    __bf16* __restrict__ qout) {
  const int n = blockIdx.x;
  const int t = threadIdx.x;
  const float* src = emb + (size_t)tok[n] * D_DIM;

  float4 a = ((const float4*)src)[t * 2 + 0];
  float4 b = ((const float4*)src)[t * 2 + 1];
  float ss = a.x*a.x + a.y*a.y + a.z*a.z + a.w*a.w
           + b.x*b.x + b.y*b.y + b.z*b.z + b.w*b.w;

  __shared__ float red[256];
  red[t] = ss;
  __syncthreads();
#pragma unroll
  for (int s = 128; s > 0; s >>= 1) {
    if (t < s) red[t] += red[t + s];
    __syncthreads();
  }
  const float scale = 1.0f / fmaxf(sqrtf(red[0]), 1e-12f);

  uint4 o;
  o.x = pack_bf16x2(a.x * scale, a.y * scale);
  o.y = pack_bf16x2(a.z * scale, a.w * scale);
  o.z = pack_bf16x2(b.x * scale, b.y * scale);
  o.w = pack_bf16x2(b.z * scale, b.w * scale);
  ((uint4*)(qout + (size_t)n * D_DIM))[t] = o;   // 8 bf16 per thread
}

// ---------------------------------------------------------------------------
// Kernel 2: bf16 WMMA GEMM (fused fp32->bf16 of emb through LDS) + top-5 over
// the query axis per vocab column.
// ---------------------------------------------------------------------------
__global__ void __launch_bounds__(256, 1)
tk_knn(const float* __restrict__ emb, const __bf16* __restrict__ qbf,
       int* __restrict__ out_idx, float* __restrict__ out_val) {
  __shared__ __align__(16) unsigned char lds_raw[LDS_BYTES];
  __bf16* ldsB = (__bf16*)lds_raw;

  const int tid  = threadIdx.x;
  const int lane = tid & 31;
  const int wid  = tid >> 5;
  const int v0   = blockIdx.x * COLS_PER_WG;

  v8f acc[MBLOCKS][NTILES];
#pragma unroll
  for (int mb = 0; mb < MBLOCKS; ++mb)
#pragma unroll
    for (int t = 0; t < NTILES; ++t)
      acc[mb][t] = (v8f){0.f, 0.f, 0.f, 0.f, 0.f, 0.f, 0.f, 0.f};

  const int am  = lane & 15;          // A row within m-block / B column
  const int ahi = (lane >> 4) * 8;    // A K sub-offset for high half-wave
  const int bhi = (lane >> 4) * 8;    // B K sub-offset for high half-wave

  for (int kc = 0; kc < D_DIM; kc += KCHUNK) {
    __syncthreads();   // previous chunk fully consumed before overwrite
    // ---- stage emb[v0..v0+63][kc..kc+KCHUNK) as bf16 into LDS ----
#pragma unroll
    for (int i = 0; i < 16; ++i) {
      const int e4  = tid + i * 256;       // 4096 float4s = 64 rows x 64 float4
      const int row = e4 >> 6;
      const int c4  = e4 & 63;
      const float4 f = *(const float4*)(emb + (size_t)(v0 + row) * D_DIM + kc + c4 * 4);
      unsigned int* dst = (unsigned int*)(ldsB + row * LDS_STRIDE + c4 * 4);
      dst[0] = pack_bf16x2(f.x, f.y);
      dst[1] = pack_bf16x2(f.z, f.w);
    }
    __syncthreads();

    // speculative prefetch of next K chunk (global_prefetch_b8 path)
    if (kc + KCHUNK < D_DIM) {
#pragma unroll
      for (int i = 0; i < 2; ++i) {
        const int line = tid + i * 256;        // 512 x 128B segments
        const int row  = line >> 3;
        const int seg  = line & 7;
        __builtin_prefetch((const void*)(emb + (size_t)(v0 + row) * D_DIM
                                         + kc + KCHUNK + seg * 32), 0, 1);
      }
    }

    const __bf16* qbase = qbf + (size_t)(wid * M_PER_WAVE) * D_DIM + kc;

    for (int ks = 0; ks < KSTEPS; ++ks) {
      // B fragments: lane holds K {0..7,16..23} (+8 hi lanes) of column (lane&15)
      v16bf bfrag[NTILES];
#pragma unroll
      for (int t = 0; t < NTILES; ++t) {
        const __bf16* bp = ldsB + (t * 16 + am) * LDS_STRIDE + ks * 32 + bhi;
        union { v16bf v; v8bf h[2]; } ub;
        ub.h[0] = *(const v8bf*)(bp);
        ub.h[1] = *(const v8bf*)(bp + 16);
        bfrag[t] = ub.v;
      }
#pragma unroll
      for (int mb = 0; mb < MBLOCKS; ++mb) {
        // A fragment per ISA 16-bit 16x32 layout: K {0..7,16..23} (+8 hi lanes)
        const __bf16* ap = qbase + (size_t)(mb * 16 + am) * D_DIM + ks * 32 + ahi;
        union { v16bf v; v8bf h[2]; } ua;
        ua.h[0] = *(const v8bf*)(ap);
        ua.h[1] = *(const v8bf*)(ap + 16);
#pragma unroll
        for (int t = 0; t < NTILES; ++t) {
          acc[mb][t] = __builtin_amdgcn_wmma_f32_16x16x32_bf16(
              false, ua.v, false, bfrag[t], (short)0, acc[mb][t], false, false);
        }
      }
    }
  }

  // ---- per-lane top-5 over this wave's 64 query slots per column ----
  float tv[NTILES][TOPK];
  int   ti[NTILES][TOPK];
#pragma unroll
  for (int t = 0; t < NTILES; ++t)
#pragma unroll
    for (int s = 0; s < TOPK; ++s) { tv[t][s] = -INFINITY; ti[t][s] = 0; }

#pragma unroll
  for (int t = 0; t < NTILES; ++t)
#pragma unroll
    for (int mb = 0; mb < MBLOCKS; ++mb)
#pragma unroll
      for (int r = 0; r < 8; ++r) {
        const int qi = wid * M_PER_WAVE + mb * 16 + r + ahi;  // C: M = r (+8 hi lanes)
        insert5(tv[t], ti[t], acc[mb][t][r], qi);
      }

  // ---- cross-wave merge through LDS ----
  __syncthreads();   // done with emb tile region
  float* mval = (float*)lds_raw;                          // 64*16*5 floats
  int*   midx = (int*)(lds_raw + COLS_PER_WG * 16 * TOPK * 4);
  const int p = wid * 2 + (lane >> 4);                    // 16 partials / column
#pragma unroll
  for (int t = 0; t < NTILES; ++t) {
    const int c = t * 16 + am;
#pragma unroll
    for (int s = 0; s < TOPK; ++s) {
      mval[(c * 16 + p) * TOPK + s] = tv[t][s];
      midx[(c * 16 + p) * TOPK + s] = ti[t][s];
    }
  }
  __syncthreads();

  if (tid < COLS_PER_WG) {
    float bv[TOPK]; int bi[TOPK];
#pragma unroll
    for (int s = 0; s < TOPK; ++s) { bv[s] = -INFINITY; bi[s] = 0; }
    for (int pp = 0; pp < 16; ++pp)
#pragma unroll
      for (int s = 0; s < TOPK; ++s)
        insert5(bv, bi, mval[(tid * 16 + pp) * TOPK + s],
                        midx[(tid * 16 + pp) * TOPK + s]);
#pragma unroll
    for (int s = 0; s < TOPK; ++s) {
      out_idx[s * V_SIZE + v0 + tid] = bi[s];
      out_val[s * V_SIZE + v0 + tid] = bv[s];
    }
  }
}

// ---------------------------------------------------------------------------
extern "C" void kernel_launch(void* const* d_in, const int* in_sizes, int n_in,
                              void* d_out, int out_size, void* d_ws, size_t ws_size,
                              hipStream_t stream) {
  const float* emb = (const float*)d_in[0];   // [V, D] fp32
  const int*   tok = (const int*)d_in[1];     // [N] int32
  __bf16* qbf = (__bf16*)d_ws;                // [N, D] bf16 (4 MB scratch)

  // outputs flat in return order: top_idx [K,V] int32, then top_vals [K,V] f32
  int*   out_idx = (int*)d_out;
  float* out_val = (float*)d_out + (size_t)TOPK * V_SIZE;

  tk_gather_normalize<<<N_Q, 256, 0, stream>>>(emb, tok, qbf);
  tk_knn<<<V_SIZE / COLS_PER_WG, 256, 0, stream>>>(emb, qbf, out_idx, out_val);
}
